// ReflectanceWeighting_66649302499629
// MI455X (gfx1250) — matrix operand order repaired and checked by reading, not compiled, not run on hardware
//
#include <hip/hip_runtime.h>

typedef __attribute__((ext_vector_type(16))) _Float16 v16h;
typedef __attribute__((ext_vector_type(8)))  float    v8f;

#define HID     64
#define NUM_SEG 2048

__device__ __forceinline__ float relu1(float v) {
    // single v_med3_num_f32: clamp to [0, +inf); inputs are finite here
    return __builtin_amdgcn_fmed3f(v, 0.0f, __builtin_inff());
}

// ---------------------------------------------------------------------------
// Kernel 0: zero the pooled-sum + count workspace (graph-capture safe init).
// ---------------------------------------------------------------------------
__global__ void rw_zero_kernel(float* p, int n) {
    int i = blockIdx.x * blockDim.x + threadIdx.x;
    int stride = gridDim.x * blockDim.x;
    for (; i < n; i += stride) p[i] = 0.0f;
}

// ---------------------------------------------------------------------------
// Kernel 1 (fused): h1 = relu(x*W1+b1) in-register (f16), h2 = relu(W2*h1+b2)
// via v_wmma_f32_16x16x32_f16, segment sums carried in registers across tiles
// (batch is sorted), flushed with a cross-lane reduction + atomics only at
// segment boundaries.
//
// Lane maps (wave32, per CDNA5 ISA):
//   A 16x32 f16 : lane l -> row M=l%16, hi=l/16;
//                 elem e<8  -> K = 8*hi + e ; elem e>=8 -> K = 16+8*hi+(e-8)
//   B 32x16 f16 : lane l -> col N=l%16, hi=l/16; elem e -> same K as A
//   C/D 16x16 f32: vgpr r, lane l -> (M = r + 8*hi, N = l%16)
// ---------------------------------------------------------------------------
__global__ void __launch_bounds__(256)
rw_mlp_pool_kernel(const float* __restrict__ x,
                   const int* __restrict__ batch_lo,  // low dwords of int64 batch
                   const float* __restrict__ W1,
                   const float* __restrict__ b1,
                   const float* __restrict__ W2,
                   const float* __restrict__ b2,
                   float* __restrict__ pooled,   // [NUM_SEG*HID]
                   float* __restrict__ counts,   // [NUM_SEG]
                   long long n)
{
    const int lane = threadIdx.x & 31;
    const int half = lane >> 4;   // hi: 0 or 1
    const int l16  = lane & 15;

    // ---- per-wave constant fragments (registers, loaded once) -------------
    v16h a[4][2];
#pragma unroll
    for (int m = 0; m < 4; ++m) {
#pragma unroll
        for (int kb = 0; kb < 2; ++kb) {
#pragma unroll
            for (int e = 0; e < 16; ++e) {
                int k = 32 * kb + ((e < 8) ? (8 * half + e)
                                           : (16 + 8 * half + (e - 8)));
                int row = 16 * m + l16;
                a[m][kb][e] = (_Float16)W2[row * HID + k];
            }
        }
    }
    v16h w1f[2], b1f[2];
#pragma unroll
    for (int kb = 0; kb < 2; ++kb) {
#pragma unroll
        for (int e = 0; e < 16; ++e) {
            int k = 32 * kb + ((e < 8) ? (8 * half + e) : (16 + 8 * half + (e - 8)));
            w1f[kb][e] = (_Float16)W1[k];
            b1f[kb][e] = (_Float16)b1[k];
        }
    }
    v8f bias[4];
#pragma unroll
    for (int m = 0; m < 4; ++m)
#pragma unroll
        for (int r = 0; r < 8; ++r)
            bias[m][r] = b2[16 * m + 8 * half + r];

    // ---- contiguous tile range for this wave ------------------------------
    long long gwave = (long long)(blockIdx.x * blockDim.x + threadIdx.x) >> 5;
    long long nwaves = ((long long)gridDim.x * blockDim.x) >> 5;
    long long tiles = (n + 15) >> 4;
    long long tpw = (tiles + nwaves - 1) / nwaves;
    long long t0 = gwave * tpw;
    long long t1 = t0 + tpw; if (t1 > tiles) t1 = tiles;

    // ---- running segment accumulator (flushed at segment boundaries) ------
    int runSeg = -1;
    int runCnt = 0;               // wave-uniform -> SGPR adds
    v8f run[4];
#pragma unroll
    for (int m = 0; m < 4; ++m)
#pragma unroll
        for (int r = 0; r < 8; ++r) run[m][r] = 0.0f;

    auto flushRun = [&]() {
        if (runSeg >= 0) {
            // xor-net reduction over the 16 points (stays inside 16-groups)
#pragma unroll
            for (int m = 0; m < 4; ++m) {
#pragma unroll
                for (int r = 0; r < 8; ++r) {
                    float v = run[m][r];
                    v += __shfl_xor(v, 1, 32);
                    v += __shfl_xor(v, 2, 32);
                    v += __shfl_xor(v, 4, 32);
                    v += __shfl_xor(v, 8, 32);
                    run[m][r] = v;   // all lanes in group hold the group sum
                }
            }
            // distribute the 32 (m,r) channel-atomics over the 16 lanes
            float* dst = pooled + (long long)runSeg * HID;
#pragma unroll
            for (int rep = 0; rep < 2; ++rep) {
                int idx = l16 + 16 * rep;      // 0..31
                int m = idx >> 3, r = idx & 7;
                atomicAdd(&dst[16 * m + 8 * half + r], run[m][r]);
            }
            if (lane == 0) atomicAdd(&counts[runSeg], (float)runCnt);
        }
#pragma unroll
        for (int m = 0; m < 4; ++m)
#pragma unroll
            for (int r = 0; r < 8; ++r) run[m][r] = 0.0f;
        runCnt = 0;
    };

    for (long long t = t0; t < t1; ++t) {
        long long base = t << 4;
        long long p = base + l16;
        bool fullTile = (base + 16 <= n);       // wave-uniform
        float xv;
        int seg;
        if (fullTile) {                         // common path: no exec masking
            xv = x[p];
            seg = batch_lo[2 * p];              // int64 low dword (< 2048)
        } else {
            bool pv = (p < n);
            xv = pv ? x[p] : 0.0f;
            seg = pv ? batch_lo[2 * p] : -1;
        }
        _Float16 xh = (_Float16)xv;

        // B fragments: relu(x*W1 + b1), packed f16
        v16h B0, B1;
#pragma unroll
        for (int e = 0; e < 16; ++e) {
            _Float16 h0 = (_Float16)(xh * w1f[0][e] + b1f[0][e]);
            _Float16 h1 = (_Float16)(xh * w1f[1][e] + b1f[1][e]);
            B0[e] = (h0 > (_Float16)0.0f) ? h0 : (_Float16)0.0f;
            B1[e] = (h1 > (_Float16)0.0f) ? h1 : (_Float16)0.0f;
        }

        // Layer 2: 8x v_wmma_f32_16x16x32_f16, f32 accumulation from bias
        v8f acc[4];
#pragma unroll
        for (int m = 0; m < 4; ++m) {
            acc[m] = __builtin_amdgcn_wmma_f32_16x16x32_f16(
                false, a[m][0], false, B0, (short)0, bias[m], false, false);
            acc[m] = __builtin_amdgcn_wmma_f32_16x16x32_f16(
                false, a[m][1], false, B1, (short)0, acc[m], false, false);
#pragma unroll
            for (int r = 0; r < 8; ++r) acc[m][r] = relu1(acc[m][r]);
        }

        // ---- segment handling ----
        int seg0 = __builtin_amdgcn_readfirstlane(seg);  // SALU, no DS
        bool uni = fullTile && (bool)__all(seg == seg0);

        if (uni) {
            if (seg0 != runSeg) {                  // wave-uniform branch
                flushRun();
                runSeg = seg0;
            }
            // steady state: 32 dual-issueable v_add_f32, no DS, no atomics
#pragma unroll
            for (int m = 0; m < 4; ++m)
#pragma unroll
                for (int r = 0; r < 8; ++r) run[m][r] += acc[m][r];
            runCnt += 16;
        } else {
            // rare: mixed segments / tail tile -> per-point atomics
            if (seg >= 0) {
#pragma unroll
                for (int m = 0; m < 4; ++m)
#pragma unroll
                    for (int r = 0; r < 8; ++r)
                        atomicAdd(&pooled[(long long)seg * HID +
                                          16 * m + 8 * half + r],
                                  acc[m][r]);
                if (half == 0) atomicAdd(&counts[seg], 1.0f);
            }
        }
    }
    flushRun();
}

// ---------------------------------------------------------------------------
// Kernel 2: per-segment mean, W3 dot, bias, relu -> weights[NUM_SEG]
// ---------------------------------------------------------------------------
__global__ void rw_finalize_kernel(const float* __restrict__ pooled,
                                   const float* __restrict__ counts,
                                   const float* __restrict__ W3,
                                   const float* __restrict__ b3,
                                   float* __restrict__ weights)
{
    int s = blockIdx.x * blockDim.x + threadIdx.x;
    if (s >= NUM_SEG) return;
    float inv = 1.0f / fmaxf(counts[s], 1.0f);
    float w = b3[0];
    const float* row = pooled + (long long)s * HID;
#pragma unroll 8
    for (int j = 0; j < HID; ++j) w += W3[j] * row[j] * inv;
    weights[s] = relu1(w);
}

// ---------------------------------------------------------------------------
// Kernel 3: out[n] = weights[batch[n]]  (read low dword of int64 batch)
// ---------------------------------------------------------------------------
__global__ void rw_gather_kernel(const int* __restrict__ batch_lo,
                                 const float* __restrict__ weights,
                                 float* __restrict__ out,
                                 long long n)
{
    long long i = (long long)blockIdx.x * blockDim.x + threadIdx.x;
    long long stride = (long long)gridDim.x * blockDim.x;
    for (; i < n; i += stride) out[i] = weights[batch_lo[2 * i]];
}

// ---------------------------------------------------------------------------
extern "C" void kernel_launch(void* const* d_in, const int* in_sizes, int n_in,
                              void* d_out, int out_size, void* d_ws, size_t ws_size,
                              hipStream_t stream)
{
    const float* x        = (const float*)d_in[0];
    const int*   batch_lo = (const int*)d_in[1];   // int64 array, LE low dwords
    const float* W1       = (const float*)d_in[2];
    const float* b1       = (const float*)d_in[3];
    const float* W2       = (const float*)d_in[4];
    const float* b2       = (const float*)d_in[5];
    const float* W3       = (const float*)d_in[6];
    const float* b3       = (const float*)d_in[7];
    float*       out      = (float*)d_out;

    long long n = (long long)in_sizes[0];   // N (IN_DIM == 1)

    // workspace layout: pooled [NUM_SEG*HID] | counts [NUM_SEG] | weights [NUM_SEG]
    float* pooled  = (float*)d_ws;
    float* counts  = pooled + (size_t)NUM_SEG * HID;
    float* weights = counts + NUM_SEG;

    // zero pooled + counts
    int zn = NUM_SEG * HID + NUM_SEG;
    rw_zero_kernel<<<(zn + 255) / 256, 256, 0, stream>>>(pooled, zn);

    // fused MLP + segment-sum: 1024 blocks x 256 thr = 8192 waves,
    // each wave owns ~16 contiguous 16-point tiles (~256 points).
    rw_mlp_pool_kernel<<<1024, 256, 0, stream>>>(
        x, batch_lo, W1, b1, W2, b2, pooled, counts, n);

    // segment finalize
    rw_finalize_kernel<<<(NUM_SEG + 255) / 256, 256, 0, stream>>>(
        pooled, counts, W3, b3, weights);

    // gather back to points
    int gblocks = (int)((n + 255) / 256);
    if (gblocks > 4096) gblocks = 4096;
    rw_gather_kernel<<<gblocks, 256, 0, stream>>>(batch_lo, weights, out, n);
}
